// PointerNetwork_86981677678930
// MI455X (gfx1250) — matrix operand order repaired and checked by reading, not compiled, not run on hardware
//
#include <hip/hip_runtime.h>
#include <hip/hip_bf16.h>

// Pointer network (encoder LSTM -> decoder LSTM + attention) for MI455X/gfx1250.
// GEMMs use V_WMMA_F32_16X16X32_BF16 (bf16 in, fp32 accumulate), 32x64 wave
// tiles, ping-pong double buffering (no register copies), fully unrolled
// k-loop with immediate-offset loads.

typedef __attribute__((ext_vector_type(16))) __bf16 v16bf;
typedef __attribute__((ext_vector_type(8)))  __bf16 bfx8;
typedef __attribute__((ext_vector_type(8)))  float  v8f;

#define BB     256   // batch
#define TT     256   // timesteps
#define FF     511   // raw features
#define UU     512   // units
#define FOURU  2048  // 4*U
#define KC     1024  // fused K = U (x) + U (h)

__device__ __forceinline__ float sigf(float x) { return 1.0f / (1.0f + __expf(-x)); }

// ---------------------------------------------------------------------------
// Pack [Wx;Wh] (each [U,4U] f32 row-major) into WT [4U, KC] bf16 (transposed,
// K-fused) so WMMA B fragments are contiguous 32B loads.
__global__ void pack_weights_kernel(const float* __restrict__ Wx,
                                    const float* __restrict__ Wh,
                                    __bf16* __restrict__ WT) {
  int idx = blockIdx.x * blockDim.x + threadIdx.x;   // FOURU*KC threads
  int k = idx & (KC - 1);
  int n = idx >> 10;
  float v = (k < UU) ? Wx[(size_t)k * FOURU + n]
                     : Wh[(size_t)(k - UU) * FOURU + n];
  WT[idx] = (__bf16)v;
}

// x = concat(inputs, score) -> bf16, layout [T, B, U] so x_t is contiguous.
__global__ void build_x_kernel(const float* __restrict__ inputs,
                               const float* __restrict__ score,
                               __bf16* __restrict__ Xbf) {
  int idx = blockIdx.x * blockDim.x + threadIdx.x;   // T*B*U threads
  int u = idx & (UU - 1);
  int r = idx >> 9;
  int b = r & (BB - 1);
  int t = r >> 8;
  float v = (u < FF) ? inputs[((size_t)b * TT + t) * FF + u] : score[b];
  Xbf[idx] = (__bf16)v;
}

// dec_in0 = mean_t x[:,t,:]  -> bf16 directly into xin buffer
__global__ void mean_x_kernel(const float* __restrict__ inputs,
                              const float* __restrict__ score,
                              __bf16* __restrict__ xin_bf) {
  int idx = blockIdx.x * blockDim.x + threadIdx.x;   // B*U threads
  int u = idx & (UU - 1);
  int b = idx >> 9;
  float m;
  if (u < FF) {
    float s = 0.0f;
    for (int t = 0; t < TT; ++t) s += inputs[((size_t)b * TT + t) * FF + u];
    m = s * (1.0f / (float)TT);
  } else {
    m = score[b];
  }
  xin_bf[idx] = (__bf16)m;
}

__global__ void zero_state_kernel(float* __restrict__ h, float* __restrict__ c,
                                  __bf16* __restrict__ hbf) {
  int idx = blockIdx.x * blockDim.x + threadIdx.x;   // B*U threads
  h[idx] = 0.0f; c[idx] = 0.0f; hbf[idx] = (__bf16)0.0f;
}

// ---------------------------------------------------------------------------
// Z[B,4U] = [A0 | A1] @ [WT]^T + bias   (K fused: A0 = x part, A1 = h part)
// A0/A1: [B, U] bf16 row-major.  WT: [4U, KC] bf16 row-major (pre-transposed).
// Block = 4 waves; each wave computes a 32(M) x 64(N) tile: 2 A-frags x
// 4 B-frags -> 8 WMMAs per 32-wide k-tile.  Two fragment sets ping-pong so
// loads for tile kt+2 are in flight while tile kt computes; fully unrolled
// k-loop turns all loads into base+immediate global_load_b128.
__global__ __launch_bounds__(128)
void lstm_gemm_kernel(const __bf16* __restrict__ A0,
                      const __bf16* __restrict__ A1,
                      const __bf16* __restrict__ WT,
                      const float* __restrict__ bias,
                      float* __restrict__ Z) {
  const int lane = threadIdx.x & 31;
  const int wave = threadIdx.x >> 5;
  const int hi   = lane >> 4;                      // half-wave select
  const int l16  = lane & 15;
  const int m0   = blockIdx.x * 32 + l16;          // 8 M-groups of 32 rows
  const int n64  = blockIdx.y * 256 + wave * 64;   // N strip base

  // Per-lane base pointers; k-loop loads use compile-time immediate offsets.
  // A fragment: lanes 0-15 K kk+0..15 contiguous, lanes 16-31 K kk+16..31
  // (folded into the base via hi*16), matching the B fragment K order.
  const __bf16* ax0 = A0 + (size_t)m0 * UU + hi * 16;
  const __bf16* ax1 = A0 + (size_t)(m0 + 16) * UU + hi * 16;
  const __bf16* ah0 = A1 + (size_t)m0 * UU + hi * 16;
  const __bf16* ah1 = A1 + (size_t)(m0 + 16) * UU + hi * 16;
  // B fragment: lane = N, 16 contiguous K halves (lanes>=16 offset +16).
  const __bf16* bp0 = WT + (size_t)(n64 +  0 + l16) * KC + hi * 16;
  const __bf16* bp1 = WT + (size_t)(n64 + 16 + l16) * KC + hi * 16;
  const __bf16* bp2 = WT + (size_t)(n64 + 32 + l16) * KC + hi * 16;
  const __bf16* bp3 = WT + (size_t)(n64 + 48 + l16) * KC + hi * 16;

  v8f acc[2][4] = {};
  v16bf aA[2], bA[4], aB[2], bB[4];

  auto load_tile = [&](int kt, v16bf* a, v16bf* b) {
    const __bf16* p0 = (kt < 16) ? ax0 : ah0;
    const __bf16* p1 = (kt < 16) ? ax1 : ah1;
    const int kk = (kt & 15) * 32;
    bfx8 l0 = *(const bfx8*)(p0 + kk);
    bfx8 h0 = *(const bfx8*)(p0 + kk + 8);
    bfx8 l1 = *(const bfx8*)(p1 + kk);
    bfx8 h1 = *(const bfx8*)(p1 + kk + 8);
#pragma unroll
    for (int i = 0; i < 8; ++i) {
      a[0][i] = l0[i]; a[0][i + 8] = h0[i];
      a[1][i] = l1[i]; a[1][i + 8] = h1[i];
    }
    const int kb = kt * 32;
    b[0] = *(const v16bf*)(bp0 + kb);
    b[1] = *(const v16bf*)(bp1 + kb);
    b[2] = *(const v16bf*)(bp2 + kb);
    b[3] = *(const v16bf*)(bp3 + kb);
  };
  auto mma_tile = [&](const v16bf* a, const v16bf* b) {
#pragma unroll
    for (int mt = 0; mt < 2; ++mt)
#pragma unroll
      for (int s = 0; s < 4; ++s)
        acc[mt][s] = __builtin_amdgcn_wmma_f32_16x16x32_bf16(
            false, a[mt], false, b[s], (short)0, acc[mt][s], false, false);
  };

  load_tile(0, aA, bA);                 // prologue: tiles 0 and 1 in flight
  load_tile(1, aB, bB);
#pragma unroll
  for (int kp = 0; kp < 16; ++kp) {     // K = 1024 as 16 pairs of 32-wide tiles
    const int kt = 2 * kp;
    mma_tile(aA, bA);
    if (kt + 2 < 32) load_tile(kt + 2, aA, bA);
    mma_tile(aB, bB);
    if (kt + 3 < 32) load_tile(kt + 3, aB, bB);
  }

  // C/D layout: VGPR r, lanes 0-15 -> M=r ; lanes 16-31 -> M=r+8 ; N=l16
#pragma unroll
  for (int mt = 0; mt < 2; ++mt)
#pragma unroll
    for (int s = 0; s < 4; ++s) {
      const int n = n64 + s * 16 + l16;
      const float bv = bias[n];
#pragma unroll
      for (int r = 0; r < 8; ++r) {
        const int row = blockIdx.x * 32 + mt * 16 + hi * 8 + r;
        Z[(size_t)row * FOURU + n] = acc[mt][s][r] + bv;
      }
    }
}

// ---------------------------------------------------------------------------
// LSTM gate nonlinearity: i,f,g,o -> h,c.  Optionally records enc_out[b,t,:].
__global__ void lstm_gates_kernel(const float* __restrict__ Z,
                                  float* __restrict__ h, float* __restrict__ c,
                                  __bf16* __restrict__ hbf,
                                  __bf16* __restrict__ enc_out, int t) {
  int idx = blockIdx.x * blockDim.x + threadIdx.x;   // B*U threads
  int u = idx & (UU - 1);
  int b = idx >> 9;
  const float* zr = Z + (size_t)b * FOURU;
  float zi = zr[u], zf = zr[u + UU], zg = zr[u + 2 * UU], zo = zr[u + 3 * UU];
  float cn = sigf(zf) * c[idx] + sigf(zi) * tanhf(zg);
  float hn = sigf(zo) * tanhf(cn);
  c[idx] = cn;
  h[idx] = hn;
  hbf[idx] = (__bf16)hn;
  if (enc_out) enc_out[((size_t)b * TT + t) * UU + u] = (__bf16)hn;
}

// ---------------------------------------------------------------------------
// Attention: scores = h . enc_out ; ptr = softmax(scores*w+b) -> out[b,s,:]
// context = ptr . enc_out -> next decoder input (bf16).
// One block per batch; enc_out [B,T,U] bf16 is L2-resident (64 MB << 192 MB L2).
__global__ __launch_bounds__(256)
void attn_kernel(const float* __restrict__ h, const __bf16* __restrict__ enc_out,
                 const float* __restrict__ ptr_w, const float* __restrict__ ptr_b,
                 float* __restrict__ out, int s, __bf16* __restrict__ xin_bf) {
  __shared__ float hs[UU];
  __shared__ float sc[TT];
  __shared__ float red[256];
  const int b = blockIdx.x;
  const int tid = threadIdx.x;

  hs[tid]       = h[(size_t)b * UU + tid];
  hs[tid + 256] = h[(size_t)b * UU + tid + 256];
  __syncthreads();

  const __bf16* eb = enc_out + (size_t)b * TT * UU;

  // score for timestep t = tid
  {
    const __bf16* row = eb + (size_t)tid * UU;
    float acc = 0.0f;
    for (int u = 0; u < UU; u += 8) {
      bfx8 e = *(const bfx8*)(row + u);
#pragma unroll
      for (int j = 0; j < 8; ++j) acc += hs[u + j] * (float)e[j];
    }
    sc[tid] = acc * ptr_w[0] + ptr_b[0];
  }
  __syncthreads();

  // softmax over 256 scores
  red[tid] = sc[tid];
  __syncthreads();
  for (int o = 128; o > 0; o >>= 1) {
    if (tid < o) red[tid] = fmaxf(red[tid], red[tid + o]);
    __syncthreads();
  }
  const float mx = red[0];
  __syncthreads();
  const float e = __expf(sc[tid] - mx);
  red[tid] = e;
  __syncthreads();
  for (int o = 128; o > 0; o >>= 1) {
    if (tid < o) red[tid] += red[tid + o];
    __syncthreads();
  }
  const float p = e * (1.0f / red[0]);
  __syncthreads();
  sc[tid] = p;                                   // reuse as ptr distribution
  out[((size_t)b * TT + s) * TT + tid] = p;
  __syncthreads();

  // context: u = tid and tid+256
  float c0 = 0.0f, c1 = 0.0f;
  for (int t2 = 0; t2 < TT; ++t2) {
    const float pv = sc[t2];
    const __bf16* row = eb + (size_t)t2 * UU;
    c0 += pv * (float)row[tid];
    c1 += pv * (float)row[tid + 256];
  }
  xin_bf[(size_t)b * UU + tid]       = (__bf16)c0;
  xin_bf[(size_t)b * UU + tid + 256] = (__bf16)c1;
}

// ---------------------------------------------------------------------------
extern "C" void kernel_launch(void* const* d_in, const int* in_sizes, int n_in,
                              void* d_out, int out_size, void* d_ws, size_t ws_size,
                              hipStream_t stream) {
  (void)in_sizes; (void)n_in; (void)out_size; (void)ws_size;
  const float* inputs = (const float*)d_in[0];
  const float* score  = (const float*)d_in[1];
  const float* enc_Wx = (const float*)d_in[2];
  const float* enc_Wh = (const float*)d_in[3];
  const float* enc_b  = (const float*)d_in[4];
  const float* dec_Wx = (const float*)d_in[5];
  const float* dec_Wh = (const float*)d_in[6];
  const float* dec_b  = (const float*)d_in[7];
  const float* ptr_w  = (const float*)d_in[8];
  const float* ptr_b  = (const float*)d_in[9];
  float* out = (float*)d_out;

  // scratch carve-out (~140 MB total)
  char* ws = (char*)d_ws;
  size_t off = 0;
  auto take = [&](size_t bytes) -> char* {
    char* p = ws + off;
    off += (bytes + 255) & ~(size_t)255;
    return p;
  };
  __bf16* WT_enc  = (__bf16*)take((size_t)FOURU * KC * 2);     //   4 MB
  __bf16* WT_dec  = (__bf16*)take((size_t)FOURU * KC * 2);     //   4 MB
  __bf16* Xbf     = (__bf16*)take((size_t)TT * BB * UU * 2);   //  64 MB
  __bf16* enc_out = (__bf16*)take((size_t)BB * TT * UU * 2);   //  64 MB
  float*  hf      = (float*) take((size_t)BB * UU * 4);        // 512 KB
  float*  cf      = (float*) take((size_t)BB * UU * 4);        // 512 KB
  __bf16* hbf     = (__bf16*)take((size_t)BB * UU * 2);        // 256 KB
  __bf16* xin_bf  = (__bf16*)take((size_t)BB * UU * 2);        // 256 KB
  float*  Zbuf    = (float*) take((size_t)BB * FOURU * 4);     //   2 MB

  // one-time (per launch) prep
  pack_weights_kernel<<<(FOURU * KC) / 256, 256, 0, stream>>>(enc_Wx, enc_Wh, WT_enc);
  pack_weights_kernel<<<(FOURU * KC) / 256, 256, 0, stream>>>(dec_Wx, dec_Wh, WT_dec);
  build_x_kernel<<<((size_t)TT * BB * UU) / 256, 256, 0, stream>>>(inputs, score, Xbf);
  mean_x_kernel<<<(BB * UU) / 256, 256, 0, stream>>>(inputs, score, xin_bf);
  zero_state_kernel<<<(BB * UU) / 256, 256, 0, stream>>>(hf, cf, hbf);

  const dim3 ggrid(8, 8);   // 8 M-groups of 32 x (8 * 4 waves * 64) N = 2048

  // encoder scan
  for (int t = 0; t < TT; ++t) {
    lstm_gemm_kernel<<<ggrid, 128, 0, stream>>>(
        Xbf + (size_t)t * BB * UU, hbf, WT_enc, enc_b, Zbuf);
    lstm_gates_kernel<<<(BB * UU) / 256, 256, 0, stream>>>(
        Zbuf, hf, cf, hbf, enc_out, t);
  }

  // decoder scan with attention (h, c carry over from encoder)
  for (int s = 0; s < TT; ++s) {
    lstm_gemm_kernel<<<ggrid, 128, 0, stream>>>(
        xin_bf, hbf, WT_dec, dec_b, Zbuf);
    lstm_gates_kernel<<<(BB * UU) / 256, 256, 0, stream>>>(
        Zbuf, hf, cf, hbf, (__bf16*)nullptr, 0);
    attn_kernel<<<BB, 256, 0, stream>>>(
        hf, enc_out, ptr_w, ptr_b, out, s, xin_bf);
  }
}
